// Attend_90589450207463
// MI455X (gfx1250) — matrix operand order repaired and checked by reading, not compiled, not run on hardware
//
#include <hip/hip_runtime.h>

typedef __attribute__((ext_vector_type(16))) _Float16 v16h;
typedef __attribute__((ext_vector_type(8)))  _Float16 v8h;
typedef __attribute__((ext_vector_type(8)))  float    v8f;

#define B_     2
#define H_     8
#define N_     2048
#define DM_    512
#define D_     64
#define CHUNK  64
#define NCHUNK (N_ / CHUNK)
#define WAVES  8
#define QBLOCK 128   // 8 waves * 16 query rows

// Flash-attention with masked_fill(-inf) + bias, f16 WMMA, f32 accumulation.
// One wave = 16 query rows; block = 128 query rows of one (b,h).
// K/V staging is double-buffered + software-pipelined through registers so
// the global loads for chunk jc+1 overlap the WMMA/softmax of chunk jc.
__global__ __launch_bounds__(256) void attend_flash_kernel(
    const float* __restrict__ qg, const float* __restrict__ kg,
    const float* __restrict__ vg, const float* __restrict__ scale_p,
    const unsigned char* __restrict__ maskg,  // jax bool -> 1 byte
    const float* __restrict__ biasg,
    float* __restrict__ outg)
{
    __shared__ _Float16 Kl[2][CHUNK * D_];       // [key][dim]     2x 8 KB
    __shared__ _Float16 Vl[2][D_ * CHUNK];       // [dim][key] (T) 2x 8 KB
    __shared__ _Float16 Pl[WAVES][16 * CHUNK];   // per-wave P       16 KB

    const int tid  = threadIdx.x;
    const int lane = tid & 31;
    const int wave = tid >> 5;
    const int ncol = lane & 15;   // N column / A-row M
    const int lh   = lane >> 4;   // lane half (0/1)
    const int kb8  = lh * 8;      // A-layout K base for this half
    const int kb16 = lh * 16;     // contiguous K base (k = 16h + s), used for A and B

    const int blk   = blockIdx.x;
    const int qt    = blk & 15;          // 16 q-tiles per (b,h)
    const int hh    = (blk >> 4) & 7;
    const int bb    = blk >> 7;
    const int qbase = qt * QBLOCK + wave * 16;

    const float scale  = scale_p[0];
    const float NEGINF = -__builtin_huge_valf();

    // ---- Q tile -> WMMA A layout (f16): lane holds row M=ncol,
    //      K elements {kb16+0..15} per 32-wide tile ----
    v16h qa0, qa1;
    {
        const int qrow = qbase + ncol;
        const float* qp = qg + ((size_t)bb * N_ + qrow) * DM_ + hh * D_;
#pragma unroll
        for (int e = 0; e < 16; ++e) {
            const int d0 = kb16 + e;
            qa0[e] = (_Float16)qp[d0];
            qa1[e] = (_Float16)qp[32 + d0];
        }
    }

    // K/V staging geometry: thread t covers elements idx = t + it*256,
    // key = idx>>6, dim = idx&63 (coalesced per instruction).
    const size_t kvbase = (size_t)bb * N_ * DM_ + hh * D_;
    float kreg[16], vreg[16];

    // ---- prologue: fetch chunk 0 and stage into buffer 0 ----
#pragma unroll
    for (int it = 0; it < 16; ++it) {
        const int idx = tid + it * 256;
        const size_t g = kvbase + (size_t)(idx >> 6) * DM_ + (idx & 63);
        kreg[it] = kg[g];
        vreg[it] = vg[g];
    }
#pragma unroll
    for (int it = 0; it < 16; ++it) {
        const int idx = tid + it * 256;
        const int key = idx >> 6, d = idx & 63;
        Kl[0][key * D_ + d]    = (_Float16)kreg[it];
        Vl[0][d * CHUNK + key] = (_Float16)vreg[it];
    }
    __syncthreads();

    v8f o0 = {}, o1 = {}, o2 = {}, o3 = {};     // O: 16x64 f32 (C layout x4)
    float m_r[8], l_r[8];
#pragma unroll
    for (int r = 0; r < 8; ++r) { m_r[r] = NEGINF; l_r[r] = 0.f; }

    _Float16* Plw = &Pl[wave][0];

    for (int jc = 0; jc < NCHUNK; ++jc) {
        const int jbase = jc * CHUNK;
        const int cur   = jc & 1;
        const bool more = (jc + 1) < NCHUNK;
        const _Float16* Kc = &Kl[cur][0];
        const _Float16* Vc = &Vl[cur][0];

        // ---- issue next chunk's K/V global loads (overlap with compute) ----
        if (more) {
            const size_t nb = kvbase + (size_t)(jbase + CHUNK) * DM_;
#pragma unroll
            for (int it = 0; it < 16; ++it) {
                const int idx = tid + it * 256;
                const size_t g = nb + (size_t)(idx >> 6) * DM_ + (idx & 63);
                kreg[it] = kg[g];
                vreg[it] = vg[g];
            }
        }

        // ---- S = Q * K^T : 4 tiles of 16x16, K-depth 64 = 2 WMMAs each ----
        v8f s[4];
#pragma unroll
        for (int jt = 0; jt < 4; ++jt) {
            const v16h b0 = *(const v16h*)&Kc[(jt * 16 + ncol) * D_ + kb16];
            const v16h b1 = *(const v16h*)&Kc[(jt * 16 + ncol) * D_ + 32 + kb16];
            v8f c = {};
            c = __builtin_amdgcn_wmma_f32_16x16x32_f16(false, qa0, false, b0,
                                                       (short)0, c, false, false);
            c = __builtin_amdgcn_wmma_f32_16x16x32_f16(false, qa1, false, b1,
                                                       (short)0, c, false, false);
            s[jt] = c;
        }

        // ---- scale, masked_fill(-inf), +bias (streamed from global) ----
        // Loads hoisted out of the select so they are UNCONDITIONAL:
        // batched global loads + v_cndmask, no divergent branches.
#pragma unroll
        for (int r = 0; r < 8; ++r) {
            const int irow = qbase + r + 8 * lh;
            const unsigned char* mrow = maskg + ((size_t)bb * N_ + irow) * N_ + jbase;
            const float*         brow = biasg + ((size_t)hh * N_ + irow) * N_ + jbase;
            // pull next chunk's bias/mask windows toward near caches
            __builtin_prefetch(brow + CHUNK, 0, 3);
            __builtin_prefetch(mrow + CHUNK, 0, 3);
            const float bv0 = brow[ 0 + ncol];
            const float bv1 = brow[16 + ncol];
            const float bv2 = brow[32 + ncol];
            const float bv3 = brow[48 + ncol];
            const unsigned char mv0 = mrow[ 0 + ncol];
            const unsigned char mv1 = mrow[16 + ncol];
            const unsigned char mv2 = mrow[32 + ncol];
            const unsigned char mv3 = mrow[48 + ncol];
            const float t0 = s[0][r] * scale + bv0;
            const float t1 = s[1][r] * scale + bv1;
            const float t2 = s[2][r] * scale + bv2;
            const float t3 = s[3][r] * scale + bv3;
            s[0][r] = mv0 ? NEGINF : t0;
            s[1][r] = mv1 ? NEGINF : t1;
            s[2][r] = mv2 ? NEGINF : t2;
            s[3][r] = mv3 ? NEGINF : t3;
        }

        // ---- online softmax: row max/sum across 16 lanes of each half ----
        float al[8];
#pragma unroll
        for (int r = 0; r < 8; ++r) {
            float t = fmaxf(fmaxf(s[0][r], s[1][r]), fmaxf(s[2][r], s[3][r]));
            t = fmaxf(t, __shfl_xor(t, 1, 32));
            t = fmaxf(t, __shfl_xor(t, 2, 32));
            t = fmaxf(t, __shfl_xor(t, 4, 32));
            t = fmaxf(t, __shfl_xor(t, 8, 32));
            const float mn = fmaxf(m_r[r], t);
            al[r] = (m_r[r] == NEGINF) ? 0.f : __expf(m_r[r] - mn);
            m_r[r] = mn;
        }
#pragma unroll
        for (int r = 0; r < 8; ++r) {
            const float mn = m_r[r];
            float ts = 0.f;
#pragma unroll
            for (int jt = 0; jt < 4; ++jt) {
                const float p = (mn == NEGINF) ? 0.f : __expf(s[jt][r] - mn);
                Plw[(r + 8 * lh) * CHUNK + jt * 16 + ncol] = (_Float16)p;
                ts += p;
            }
            ts += __shfl_xor(ts, 1, 32);
            ts += __shfl_xor(ts, 2, 32);
            ts += __shfl_xor(ts, 4, 32);
            ts += __shfl_xor(ts, 8, 32);
            l_r[r] = l_r[r] * al[r] + ts;
        }
#pragma unroll
        for (int r = 0; r < 8; ++r) {
            o0[r] *= al[r]; o1[r] *= al[r]; o2[r] *= al[r]; o3[r] *= al[r];
        }

        // P stores must land before A-layout reloads (same wave, DS in-order;
        // make it explicit and stop compiler reordering).
        asm volatile("s_wait_dscnt 0" ::: "memory");

        // ---- O += P * V : A = P (16x32 f16 x2), B = V^T chunks ----
#pragma unroll
        for (int kc = 0; kc < 2; ++kc) {
            const v8h lo = *(const v8h*)&Plw[ncol * CHUNK + kc * 32 + kb16];
            const v8h hi = *(const v8h*)&Plw[ncol * CHUNK + kc * 32 + kb16 + 8];
            v16h pa;
#pragma unroll
            for (int e = 0; e < 8; ++e) { pa[e] = lo[e]; pa[e + 8] = hi[e]; }
            const v16h vb0 = *(const v16h*)&Vc[( 0 + ncol) * CHUNK + kc * 32 + kb16];
            const v16h vb1 = *(const v16h*)&Vc[(16 + ncol) * CHUNK + kc * 32 + kb16];
            const v16h vb2 = *(const v16h*)&Vc[(32 + ncol) * CHUNK + kc * 32 + kb16];
            const v16h vb3 = *(const v16h*)&Vc[(48 + ncol) * CHUNK + kc * 32 + kb16];
            o0 = __builtin_amdgcn_wmma_f32_16x16x32_f16(false, pa, false, vb0,
                                                        (short)0, o0, false, false);
            o1 = __builtin_amdgcn_wmma_f32_16x16x32_f16(false, pa, false, vb1,
                                                        (short)0, o1, false, false);
            o2 = __builtin_amdgcn_wmma_f32_16x16x32_f16(false, pa, false, vb2,
                                                        (short)0, o2, false, false);
            o3 = __builtin_amdgcn_wmma_f32_16x16x32_f16(false, pa, false, vb3,
                                                        (short)0, o3, false, false);
        }

        // ---- stage next chunk into the other LDS buffer ----
        if (more) {
            _Float16* Kn = &Kl[cur ^ 1][0];
            _Float16* Vn = &Vl[cur ^ 1][0];
#pragma unroll
            for (int it = 0; it < 16; ++it) {
                const int idx = tid + it * 256;
                const int key = idx >> 6, d = idx & 63;
                Kn[key * D_ + d]    = (_Float16)kreg[it];
                Vn[d * CHUNK + key] = (_Float16)vreg[it];
            }
        }
        __syncthreads();
    }

    // ---- epilogue: out = O / l (0 for fully masked rows, == nan_to_num) ----
    float* obase = outg + (((size_t)bb * H_ + hh) * N_) * D_;
#pragma unroll
    for (int r = 0; r < 8; ++r) {
        const int irow = qbase + r + 8 * lh;
        const float inv = (l_r[r] > 0.f) ? (1.0f / l_r[r]) : 0.f;
        float* op = obase + (size_t)irow * D_;
        op[ 0 + ncol] = o0[r] * inv;
        op[16 + ncol] = o1[r] * inv;
        op[32 + ncol] = o2[r] * inv;
        op[48 + ncol] = o3[r] * inv;
    }
}

extern "C" void kernel_launch(void* const* d_in, const int* in_sizes, int n_in,
                              void* d_out, int out_size, void* d_ws, size_t ws_size,
                              hipStream_t stream) {
    const float*         q     = (const float*)d_in[0];
    const float*         k     = (const float*)d_in[1];
    const float*         v     = (const float*)d_in[2];
    const float*         scale = (const float*)d_in[3];
    const unsigned char* mask  = (const unsigned char*)d_in[4];  // bool (1B)
    const float*         bias  = (const float*)d_in[5];
    float*               out   = (float*)d_out;

    const int blocks = B_ * H_ * (N_ / QBLOCK);  // 2*8*16 = 256
    attend_flash_kernel<<<blocks, 256, 0, stream>>>(q, k, v, scale, mask, bias, out);
}
